// GeneratorNetwork_6296422056639
// MI455X (gfx1250) — matrix operand order repaired and checked by reading, not compiled, not run on hardware
//
#include <hip/hip_runtime.h>
#include <stdint.h>

// ---------------------------------------------------------------------------
// Types for CDNA5 WMMA
// ---------------------------------------------------------------------------
typedef __attribute__((ext_vector_type(16))) __bf16 v16bf;
typedef __attribute__((ext_vector_type(8)))  float  v8f;

#define LATENT 128
#define RNN    512
#define CODE   2048
#define G4     (4 * RNN)          // 2048
#define K1     (CODE + RNN)       // 2560  (x | h1)
#define K2     (2 * RNN)          // 1024  (h1 | h2)

// ---------------------------------------------------------------------------
// Helpers
// ---------------------------------------------------------------------------
__device__ __forceinline__ unsigned short f2bf(float f) {
    unsigned int u = __float_as_uint(f);
    u += 0x7FFFu + ((u >> 16) & 1u);          // round-to-nearest-even
    return (unsigned short)(u >> 16);
}
__device__ __forceinline__ float sigm(float x) {
    return 1.f / (1.f + __expf(-x));
}
__device__ __forceinline__ float tanh_fast(float x) {
    float e = __expf(2.f * x);
    return 1.f - 2.f / (e + 1.f);
}

union FragBF {
    uint4 u[2];
    v16bf v;
};

__device__ __forceinline__ v16bf load_frag(const unsigned short* rowp, int k0, int k1) {
    FragBF f;
    f.u[0] = *(const uint4*)(rowp + k0);
    f.u[1] = *(const uint4*)(rowp + k1);
    return f.v;
}

// ---------------------------------------------------------------------------
// bf16 WMMA GEMM:  C[M,N] = A[M,K] * Bt[N,K]^T (+bias)
//   mode 0: C = acc + bias                      (fp32 store)
//   mode 2: Xout = bf16( sigmoid(acc+bias) * idea )   (fused attention gate)
// Block: 256 thr = 8 waves; block tile 128x128; wave tile 64x32 (4x2 WMMA).
// Requires M%128==0, N%128==0, K%32==0.
// ---------------------------------------------------------------------------
#define LDSROW 40   // 32 k-elems + 8 pad (80B row => 16B-aligned, conflict-light)

__global__ __launch_bounds__(256)
void gemm_bf16_wmma(const unsigned short* __restrict__ A, int lda,
                    const unsigned short* __restrict__ Bt, int K,
                    const float* __restrict__ bias,
                    float* __restrict__ C, int ldc,
                    const float* __restrict__ idea, int ldi,
                    unsigned short* __restrict__ Xout, int ldx,
                    int mode)
{
    __shared__ unsigned short As[2][128 * LDSROW];
    __shared__ unsigned short Bs[2][128 * LDSROW];

    const int t    = threadIdx.x;
    const int bn   = blockIdx.x;
    const int bm   = blockIdx.y;
    const int wave = t >> 5;
    const int lane = t & 31;
    const int wm   = wave >> 2;     // 0..1   (64 rows each)
    const int wn   = wave & 3;      // 0..3   (32 cols each)
    const int l16  = lane & 15;
    const int half = lane >> 4;

    const unsigned short* gA = A  + (size_t)(bm * 128) * lda;
    const unsigned short* gB = Bt + (size_t)(bn * 128) * K;

    v8f acc[4][2];
#pragma unroll
    for (int mt = 0; mt < 4; ++mt)
#pragma unroll
        for (int nt = 0; nt < 2; ++nt)
            acc[mt][nt] = (v8f){0.f, 0.f, 0.f, 0.f, 0.f, 0.f, 0.f, 0.f};

    const int c0 = t, c1 = t + 256;           // 512 16B-chunks per 128x32 tile
    const int r0 = c0 >> 2, cc0 = (c0 & 3) * 8;
    const int r1 = c1 >> 2, cc1 = (c1 & 3) * 8;

    uint4 a0, a1, b0, b1;
    // prologue: stage 0
    a0 = *(const uint4*)(gA + (size_t)r0 * lda + cc0);
    a1 = *(const uint4*)(gA + (size_t)r1 * lda + cc1);
    b0 = *(const uint4*)(gB + (size_t)r0 * K + cc0);
    b1 = *(const uint4*)(gB + (size_t)r1 * K + cc1);
    *(uint4*)(&As[0][r0 * LDSROW + cc0]) = a0;
    *(uint4*)(&As[0][r1 * LDSROW + cc1]) = a1;
    *(uint4*)(&Bs[0][r0 * LDSROW + cc0]) = b0;
    *(uint4*)(&Bs[0][r1 * LDSROW + cc1]) = b1;
    __syncthreads();

    const int nk  = K >> 5;
    const int ka0 = half * 8, ka1 = 16 + half * 8;   // A frag: ISA 16-bit 16x32 layout
    const int kb0 = half * 8, kb1 = 16 + half * 8;   // B frag: ISA 16-bit 16x32 layout

    for (int kk = 0; kk < nk; ++kk) {
        const int cur = kk & 1;
        if (kk + 1 < nk) {      // prefetch next stage into regs (hidden under WMMA)
            const int kp = (kk + 1) << 5;
            a0 = *(const uint4*)(gA + (size_t)r0 * lda + kp + cc0);
            a1 = *(const uint4*)(gA + (size_t)r1 * lda + kp + cc1);
            b0 = *(const uint4*)(gB + (size_t)r0 * K + kp + cc0);
            b1 = *(const uint4*)(gB + (size_t)r1 * K + kp + cc1);
        }

        v16bf af[4], bfv[2];
#pragma unroll
        for (int mt = 0; mt < 4; ++mt)
            af[mt] = load_frag(&As[cur][(wm * 64 + mt * 16 + l16) * LDSROW], ka0, ka1);
#pragma unroll
        for (int nt = 0; nt < 2; ++nt)
            bfv[nt] = load_frag(&Bs[cur][(wn * 32 + nt * 16 + l16) * LDSROW], kb0, kb1);

#pragma unroll
        for (int mt = 0; mt < 4; ++mt)
#pragma unroll
            for (int nt = 0; nt < 2; ++nt)
                acc[mt][nt] = __builtin_amdgcn_wmma_f32_16x16x32_bf16(
                    false, af[mt], false, bfv[nt], (short)0, acc[mt][nt], false, false);

        if (kk + 1 < nk) {
            const int nxt = cur ^ 1;
            *(uint4*)(&As[nxt][r0 * LDSROW + cc0]) = a0;
            *(uint4*)(&As[nxt][r1 * LDSROW + cc1]) = a1;
            *(uint4*)(&Bs[nxt][r0 * LDSROW + cc0]) = b0;
            *(uint4*)(&Bs[nxt][r1 * LDSROW + cc1]) = b1;
        }
        __syncthreads();
    }

    // epilogue: C layout -> VGPR g, lanes 0-15: M=g, N=l16; lanes 16-31: M=g+8
    const int rowBase = bm * 128 + wm * 64;
    const int colBase = bn * 128 + wn * 32;
#pragma unroll
    for (int mt = 0; mt < 4; ++mt) {
#pragma unroll
        for (int nt = 0; nt < 2; ++nt) {
            const int col = colBase + nt * 16 + l16;
            const float bv = bias ? bias[col] : 0.f;
#pragma unroll
            for (int g = 0; g < 8; ++g) {
                const int row = rowBase + mt * 16 + half * 8 + g;
                const float v = acc[mt][nt][g] + bv;
                if (mode == 0) {
                    C[(size_t)row * ldc + col] = v;
                } else {
                    const float s = sigm(v);
                    Xout[(size_t)row * ldx + col] =
                        f2bf(s * idea[(size_t)row * ldi + col]);
                }
            }
        }
    }
}

// ---------------------------------------------------------------------------
// idea = tanh(latent @ w_unpack + b_unpack); also seed A1 x-part (att0 == 1)
// ---------------------------------------------------------------------------
__global__ __launch_bounds__(256)
void unpack_kernel(const float* __restrict__ latent,
                   const float* __restrict__ w_unpack,
                   const float* __restrict__ b_unpack,
                   float* __restrict__ idea,
                   unsigned short* __restrict__ A1)
{
    const int idx = blockIdx.x * 256 + threadIdx.x;   // B*CODE threads
    const int col = idx & (CODE - 1);
    const int b   = idx >> 11;
    float acc = b_unpack[col];
#pragma unroll 8
    for (int k = 0; k < LATENT; ++k)
        acc += latent[b * LATENT + k] * w_unpack[k * CODE + col];
    const float v = tanh_fast(acc);
    idea[idx] = v;
    A1[(size_t)b * K1 + col] = f2bf(v);
}

// Wt[n*K + k] = bf16( k<Ka ? Wa[k*N+n] : Wb[(k-Ka)*N+n] ),  N == CODE
__global__ __launch_bounds__(256)
void pack_pair_kernel(const float* __restrict__ Wa, int Ka,
                      const float* __restrict__ Wb, int Kb,
                      unsigned short* __restrict__ Wt)
{
    const int K = Ka + Kb;
    const int idx = blockIdx.x * 256 + threadIdx.x;
    if (idx >= CODE * K) return;
    const int n = idx / K, k = idx - n * K;
    const float w = (k < Ka) ? Wa[(size_t)k * CODE + n] : Wb[(size_t)(k - Ka) * CODE + n];
    Wt[(size_t)n * K + k] = f2bf(w);
}

__global__ __launch_bounds__(256)
void bias_fold_kernel(const float* a1, const float* b1, float* o1,
                      const float* a2, const float* b2, float* o2)
{
    const int i = blockIdx.x * 256 + threadIdx.x;     // G4 threads
    o1[i] = a1[i] + b1[i];
    o2[i] = a2[i] + b2[i];
}

__global__ __launch_bounds__(256)
void zero_u32_kernel(unsigned int* p, int n)
{
    const int i = blockIdx.x * 256 + threadIdx.x;
    if (i < n) p[i] = 0u;
}

// ---------------------------------------------------------------------------
// LSTM pointwise: i,f,g,o gate order; writes h (bf16) into up to two
// strided destinations (the concatenated GEMM-A buffers) + optional fp32 h.
// ---------------------------------------------------------------------------
__global__ __launch_bounds__(256)
void lstm_cell_kernel(const float* __restrict__ gates,
                      float* __restrict__ c,
                      unsigned short* __restrict__ hb0, int ld0,
                      unsigned short* __restrict__ hb1, int ld1,
                      float* __restrict__ hf)
{
    const int idx = blockIdx.x * 256 + threadIdx.x;   // B*RNN threads
    const int j = idx & (RNN - 1);
    const int b = idx >> 9;
    const float* g = gates + (size_t)b * G4;
    const float gi = sigm(g[j]);
    const float gf = sigm(g[RNN + j]);
    const float gg = tanh_fast(g[2 * RNN + j]);
    const float go = sigm(g[3 * RNN + j]);
    const float cn = gf * c[idx] + gi * gg;
    c[idx] = cn;
    const float h = go * tanh_fast(cn);
    const unsigned short hb = f2bf(h);
    if (hb0) hb0[(size_t)b * ld0 + j] = hb;
    if (hb1) hb1[(size_t)b * ld1 + j] = hb;
    if (hf)  hf[idx] = h;
}

// angles/widths heads: per-batch GEMV over RNN
__global__ __launch_bounds__(256)
void heads_kernel(const float* __restrict__ h2,
                  const float* __restrict__ w_ang, const float* __restrict__ b_ang,
                  const float* __restrict__ w_wid, const float* __restrict__ b_wid,
                  float* __restrict__ out, int T, int t)
{
    const int b = blockIdx.x * 256 + threadIdx.x;     // B threads
    const float* h = h2 + (size_t)b * RNN;
    float sa = b_ang[0], sw = b_wid[0];
#pragma unroll 8
    for (int k = 0; k < RNN; ++k) {
        const float hv = h[k];
        sa += hv * w_ang[k];
        sw += hv * w_wid[k];
    }
    float* o = out + ((size_t)b * T + t) * 2;
    o[0] = tanh_fast(sa);
    o[1] = sigm(sw);
}

// ---------------------------------------------------------------------------
// Host-side launcher
// ---------------------------------------------------------------------------
extern "C" void kernel_launch(void* const* d_in, const int* in_sizes, int n_in,
                              void* d_out, int out_size, void* d_ws, size_t ws_size,
                              hipStream_t stream)
{
    const float* latent   = (const float*)d_in[0];
    const float* w_unpack = (const float*)d_in[2];
    const float* b_unpack = (const float*)d_in[3];
    const float* w_ih1    = (const float*)d_in[4];
    const float* w_hh1    = (const float*)d_in[5];
    const float* b_ih1    = (const float*)d_in[6];
    const float* b_hh1    = (const float*)d_in[7];
    const float* w_ih2    = (const float*)d_in[8];
    const float* w_hh2    = (const float*)d_in[9];
    const float* b_ih2    = (const float*)d_in[10];
    const float* b_hh2    = (const float*)d_in[11];
    const float* w_att    = (const float*)d_in[12];
    const float* b_att    = (const float*)d_in[13];
    const float* w_wid    = (const float*)d_in[14];
    const float* b_wid    = (const float*)d_in[15];
    const float* w_ang    = (const float*)d_in[16];
    const float* b_ang    = (const float*)d_in[17];
    float* out = (float*)d_out;

    const int B = in_sizes[0] / LATENT;               // 1024
    const int T = out_size / (2 * B);                 // 128 (device scalar mirrored via out_size)

    // ---- workspace carve (all sizes 256B-multiples) ----
    uint8_t* w = (uint8_t*)d_ws;
    unsigned short* A1  = (unsigned short*)w; w += (size_t)B * K1 * 2;      // [x | h1] bf16
    unsigned short* A2  = (unsigned short*)w; w += (size_t)B * K2 * 2;      // [h1 | h2] bf16
    float* c1           = (float*)w;          w += (size_t)B * RNN * 4;
    float* c2           = (float*)w;          w += (size_t)B * RNN * 4;
    float* h2f          = (float*)w;          w += (size_t)B * RNN * 4;
    float* gates        = (float*)w;          w += (size_t)B * G4 * 4;      // shared gates buffer
    float* idea         = (float*)w;          w += (size_t)B * CODE * 4;
    unsigned short* W1t = (unsigned short*)w; w += (size_t)CODE * K1 * 2;   // [N][K]
    unsigned short* W2t = (unsigned short*)w; w += (size_t)CODE * K2 * 2;
    unsigned short* WAt = (unsigned short*)w; w += (size_t)CODE * RNN * 2;
    float* bias1        = (float*)w;          w += 8192;
    float* bias2        = (float*)w;          w += 8192;
    (void)ws_size; (void)n_in;

    // ---- preprocessing (per call; deterministic) ----
    {   // zero A1 + A2 + c1 + c2 (contiguous span)
        const int nz = (int)(((size_t)B * K1 * 2 + (size_t)B * K2 * 2 +
                              (size_t)B * RNN * 4 + (size_t)B * RNN * 4) / 4);
        zero_u32_kernel<<<(nz + 255) / 256, 256, 0, stream>>>((unsigned int*)d_ws, nz);
    }
    unpack_kernel<<<(B * CODE) / 256, 256, 0, stream>>>(latent, w_unpack, b_unpack, idea, A1);
    pack_pair_kernel<<<(CODE * K1 + 255) / 256, 256, 0, stream>>>(w_ih1, CODE, w_hh1, RNN, W1t);
    pack_pair_kernel<<<(CODE * K2 + 255) / 256, 256, 0, stream>>>(w_ih2, RNN, w_hh2, RNN, W2t);
    pack_pair_kernel<<<(CODE * RNN + 255) / 256, 256, 0, stream>>>(w_att, RNN, nullptr, 0, WAt);
    bias_fold_kernel<<<G4 / 256, 256, 0, stream>>>(b_ih1, b_hh1, bias1, b_ih2, b_hh2, bias2);

    const dim3 gemmGrid(CODE / 128, B / 128);         // (16, 8)
    const int cellBlocks = (B * RNN) / 256;

    for (int t = 0; t < T; ++t) {
        // gates1 = [x|h1] @ W1t^T + bias1
        gemm_bf16_wmma<<<gemmGrid, 256, 0, stream>>>(
            A1, K1, W1t, K1, bias1, gates, G4, nullptr, 0, nullptr, 0, 0);
        // h1,c1 update; h1(bf16) -> A1[:,2048:] and A2[:,0:512]
        lstm_cell_kernel<<<cellBlocks, 256, 0, stream>>>(
            gates, c1, A1 + CODE, K1, A2, K2, nullptr);
        // gates2 = [h1|h2] @ W2t^T + bias2
        gemm_bf16_wmma<<<gemmGrid, 256, 0, stream>>>(
            A2, K2, W2t, K2, bias2, gates, G4, nullptr, 0, nullptr, 0, 0);
        // h2,c2 update; h2(bf16) -> A2[:,512:], h2(f32) for heads
        lstm_cell_kernel<<<cellBlocks, 256, 0, stream>>>(
            gates, c2, A2 + RNN, K2, nullptr, 0, h2f);
        // angles / widths
        heads_kernel<<<B / 256, 256, 0, stream>>>(h2f, w_ang, b_ang, w_wid, b_wid, out, T, t);
        // att = sigmoid(h2 @ WAt^T + b_att); fused x = bf16(att * idea) -> A1 x-part
        if (t + 1 < T) {
            gemm_bf16_wmma<<<gemmGrid, 256, 0, stream>>>(
                A2 + RNN, K2, WAt, RNN, b_att, nullptr, 0, idea, CODE, A1, K1, 2);
        }
    }
}
